// Controller_60662118089467
// MI455X (gfx1250) — matrix operand order repaired and checked by reading, not compiled, not run on hardware
//
#include <hip/hip_runtime.h>
#include <hip/hip_bf16.h>

// ---------------------------------------------------------------------------
// NAS LSTM controller on gfx1250 (MI455X).
//  - bf16 weight cache in d_ws (134MB, fits 192MB L2) feeding
//    v_wmma_f32_16x16x32_bf16 matvecs (vector broadcast into all 16 A rows)
//  - 2 N-tiles per wave, software-pipelined: next iteration's tiles load
//    while current WMMAs execute (no full loadcnt drains per wmma)
//  - deterministic K-split (8) partial sums, reduced in the cell kernel
//  - threefry2x32 / Gumbel-argmax sampling matching JAX semantics
// ---------------------------------------------------------------------------

#define H      2048
#define G4H    8192
#define NL     2
#define STEPS  24
#define NA     16
#define KSPLIT 8
#define KSLICE (H / KSPLIT)   // 256

typedef __attribute__((ext_vector_type(16))) __bf16 v16bf;
typedef __attribute__((ext_vector_type(8)))  __bf16 v8bf;
typedef __attribute__((ext_vector_type(8)))  float  v8f;

__device__ __forceinline__ unsigned short f32_to_bf16(float f) {
  unsigned int u = __float_as_uint(f);
  unsigned int r = u + 0x7fffu + ((u >> 16) & 1u);   // round-to-nearest-even
  return (unsigned short)(r >> 16);
}

// ------------------------- fp32 -> bf16 weight cache ------------------------
__global__ __launch_bounds__(256) void cvt_f32_bf16(const float* __restrict__ src,
                                                    unsigned short* __restrict__ dst,
                                                    long long n4) {
  long long i = (long long)blockIdx.x * blockDim.x + threadIdx.x;
  long long stride = (long long)gridDim.x * blockDim.x;
  const float4* s4 = (const float4*)src;
  ushort4* d4 = (ushort4*)dst;
  for (; i < n4; i += stride) {
    float4 v = s4[i];
    ushort4 o;
    o.x = f32_to_bf16(v.x); o.y = f32_to_bf16(v.y);
    o.z = f32_to_bf16(v.z); o.w = f32_to_bf16(v.w);
    d4[i] = o;
  }
}

// ------------------------------ init state ---------------------------------
__global__ __launch_bounds__(256) void init_state(const float* __restrict__ g_emb,
                                                  unsigned short* __restrict__ x_bf,
                                                  unsigned short* __restrict__ h_bf,
                                                  float* __restrict__ h_f,
                                                  float* __restrict__ c_f) {
  int j = blockIdx.x * blockDim.x + threadIdx.x;   // 0..H-1
  x_bf[j] = f32_to_bf16(g_emb[j]);
  for (int l = 0; l < NL; ++l) {
    h_bf[l * H + j] = 0;     // bf16 +0.0
    h_f [l * H + j] = 0.f;
    c_f [l * H + j] = 0.f;
  }
}

// ------------------------ WMMA gate matvec ---------------------------------
// Per wave: 32 gate outputs (two 16-wide N tiles sharing the A fetch) over a
// 256-wide K slice:  gates += W_ih @ x + W_hh @ h.
// A: activation chunk replicated into all 16 rows (all D rows equal); 16-bit A
//    layout used here: lane-half 0 holds K 0..15; half 1 holds K 16..31.
// B: lane n (0..15) holds 16 contiguous K of weight row (32B load); lanes
//    16..31 hold K+16 of the same rows.

template<bool CACHED>
__device__ __forceinline__ void load_tiles(
    const unsigned short* __restrict__ x_bf,   const unsigned short* __restrict__ h_bf,
    const unsigned short* __restrict__ wih_r0, const unsigned short* __restrict__ wih_r1,
    const unsigned short* __restrict__ whh_r0, const unsigned short* __restrict__ whh_r1,
    const float* __restrict__ wihf_r0, const float* __restrict__ wihf_r1,
    const float* __restrict__ whhf_r0, const float* __restrict__ whhf_r1,
    int k0, int khalf,
    v16bf& ax, v16bf& ah, v16bf& bih0, v16bf& bih1, v16bf& bhh0, v16bf& bhh1)
{
  const int axoff = k0 + khalf * 16;
  v8bf xlo = *(const v8bf*)(x_bf + axoff);
  v8bf xhi = *(const v8bf*)(x_bf + axoff + 8);
  v8bf hlo = *(const v8bf*)(h_bf + axoff);
  v8bf hhi = *(const v8bf*)(h_bf + axoff + 8);
  ax = __builtin_shufflevector(xlo, xhi, 0,1,2,3,4,5,6,7,8,9,10,11,12,13,14,15);
  ah = __builtin_shufflevector(hlo, hhi, 0,1,2,3,4,5,6,7,8,9,10,11,12,13,14,15);

  const int boff = k0 + khalf * 16;
  if constexpr (CACHED) {
    bih0 = *(const v16bf*)(wih_r0 + boff);
    bih1 = *(const v16bf*)(wih_r1 + boff);
    bhh0 = *(const v16bf*)(whh_r0 + boff);
    bhh1 = *(const v16bf*)(whh_r1 + boff);
  } else {
    const float* p0 = wihf_r0 + boff; const float* p1 = wihf_r1 + boff;
    const float* q0 = whhf_r0 + boff; const float* q1 = whhf_r1 + boff;
    #pragma unroll
    for (int e = 0; e < 16; ++e) {
      union { unsigned short u; __bf16 b; } c0, c1, c2, c3;
      c0.u = f32_to_bf16(p0[e]); c1.u = f32_to_bf16(p1[e]);
      c2.u = f32_to_bf16(q0[e]); c3.u = f32_to_bf16(q1[e]);
      bih0[e] = c0.b; bih1[e] = c1.b; bhh0[e] = c2.b; bhh1[e] = c3.b;
    }
  }
}

template<bool CACHED>
__global__ __launch_bounds__(256) void gemv_gates(
    const unsigned short* __restrict__ wih_bf, const unsigned short* __restrict__ whh_bf,
    const float* __restrict__ wih_f,           const float* __restrict__ whh_f,
    const unsigned short* __restrict__ x_bf,   const unsigned short* __restrict__ h_bf,
    float* __restrict__ gates_part)
{
  const int wave  = threadIdx.x >> 5;
  const int lane  = threadIdx.x & 31;
  const int n     = lane & 15;
  const int khalf = lane >> 4;
  const int jbase = blockIdx.x * 256 + wave * 32;       // first of 32 output rows
  const int kbase = blockIdx.y * KSLICE;
  const int kend  = kbase + KSLICE;

  const unsigned short* wih_r0 = nullptr; const unsigned short* wih_r1 = nullptr;
  const unsigned short* whh_r0 = nullptr; const unsigned short* whh_r1 = nullptr;
  const float* wihf_r0 = nullptr; const float* wihf_r1 = nullptr;
  const float* whhf_r0 = nullptr; const float* whhf_r1 = nullptr;
  if constexpr (CACHED) {
    wih_r0 = wih_bf + (size_t)(jbase + n) * H;
    wih_r1 = wih_bf + (size_t)(jbase + 16 + n) * H;
    whh_r0 = whh_bf + (size_t)(jbase + n) * H;
    whh_r1 = whh_bf + (size_t)(jbase + 16 + n) * H;
  } else {
    wihf_r0 = wih_f + (size_t)(jbase + n) * H;
    wihf_r1 = wih_f + (size_t)(jbase + 16 + n) * H;
    whhf_r0 = whh_f + (size_t)(jbase + n) * H;
    whhf_r1 = whh_f + (size_t)(jbase + 16 + n) * H;
  }

  v8f acc0 = {};
  v8f acc1 = {};

  // software pipeline: tiles for iteration i+1 load while WMMAs consume i
  v16bf ax, ah, bih0, bih1, bhh0, bhh1;
  load_tiles<CACHED>(x_bf, h_bf, wih_r0, wih_r1, whh_r0, whh_r1,
                     wihf_r0, wihf_r1, whhf_r0, whhf_r1, kbase, khalf,
                     ax, ah, bih0, bih1, bhh0, bhh1);

  #pragma unroll 4
  for (int k0 = kbase; k0 < kend; k0 += 32) {
    const int kn = (k0 + 32 < kend) ? (k0 + 32) : kbase;   // clamp: no OOB on last iter
    v16bf axn, ahn, bihn0, bihn1, bhhn0, bhhn1;
    load_tiles<CACHED>(x_bf, h_bf, wih_r0, wih_r1, whh_r0, whh_r1,
                       wihf_r0, wihf_r1, whhf_r0, whhf_r1, kn, khalf,
                       axn, ahn, bihn0, bihn1, bhhn0, bhhn1);

    acc0 = __builtin_amdgcn_wmma_f32_16x16x32_bf16(false, ax, false, bih0, (short)0, acc0, false, false);
    acc1 = __builtin_amdgcn_wmma_f32_16x16x32_bf16(false, ax, false, bih1, (short)0, acc1, false, false);
    acc0 = __builtin_amdgcn_wmma_f32_16x16x32_bf16(false, ah, false, bhh0, (short)0, acc0, false, false);
    acc1 = __builtin_amdgcn_wmma_f32_16x16x32_bf16(false, ah, false, bhh1, (short)0, acc1, false, false);

    ax = axn; ah = ahn;
    bih0 = bihn0; bih1 = bihn1; bhh0 = bhhn0; bhh1 = bhhn1;
  }

  // All 16 D rows identical; lane<16 holds (M=0, N=lane) in acc[0].
  if (lane < 16) {
    float* dst = gates_part + (size_t)blockIdx.y * G4H + jbase;
    dst[lane]      = acc0[0];
    dst[16 + lane] = acc1[0];
  }
}

// --------------------------- LSTM cell update -------------------------------
__global__ __launch_bounds__(256) void lstm_cell(
    const float* __restrict__ gates_part,
    const float* __restrict__ b_ih, const float* __restrict__ b_hh,
    float* __restrict__ c_f, float* __restrict__ h_f,
    unsigned short* __restrict__ h_bf)
{
  int j = blockIdx.x * blockDim.x + threadIdx.x;   // 0..H-1
  float g4[4];
  #pragma unroll
  for (int q = 0; q < 4; ++q) {                    // gate order: i, f, g, o
    int idx = q * H + j;
    float s = b_ih[idx] + b_hh[idx];
    #pragma unroll
    for (int p = 0; p < KSPLIT; ++p) s += gates_part[p * G4H + idx];
    g4[q] = s;
  }
  float ig = 1.0f / (1.0f + expf(-g4[0]));
  float fg = 1.0f / (1.0f + expf(-g4[1]));
  float og = 1.0f / (1.0f + expf(-g4[3]));
  float cn = fg * c_f[j] + ig * tanhf(g4[2]);
  float hn = og * tanhf(cn);
  c_f[j] = cn;
  h_f[j] = hn;
  h_bf[j] = f32_to_bf16(hn);
}

// --------------------- threefry2x32 (JAX-compatible core) -------------------
__device__ __forceinline__ void threefry2x32(unsigned k0, unsigned k1,
                                             unsigned x0, unsigned x1,
                                             unsigned& o0, unsigned& o1) {
  unsigned ks0 = k0, ks1 = k1, ks2 = k0 ^ k1 ^ 0x1BD11BDAu;
  unsigned a = x0 + ks0, b = x1 + ks1;
#define TF_R(R) { a += b; b = ((b << (R)) | (b >> (32 - (R)))) ^ a; }
  TF_R(13) TF_R(15) TF_R(26) TF_R(6)   a += ks1; b += ks2 + 1u;
  TF_R(17) TF_R(29) TF_R(16) TF_R(24)  a += ks2; b += ks0 + 2u;
  TF_R(13) TF_R(15) TF_R(26) TF_R(6)   a += ks0; b += ks1 + 3u;
  TF_R(17) TF_R(29) TF_R(16) TF_R(24)  a += ks1; b += ks2 + 4u;
  TF_R(13) TF_R(15) TF_R(26) TF_R(6)   a += ks2; b += ks0 + 5u;
#undef TF_R
  o0 = a; o1 = b;
}

// ---------------- logits + categorical sample + next embedding --------------
__global__ __launch_bounds__(256) void sample_step(
    const float* __restrict__ h2, const float* __restrict__ soft_w,
    const float* __restrict__ w_emb, int step,
    unsigned short* __restrict__ x_bf, float* __restrict__ out)
{
  __shared__ float red[256];
  __shared__ int act_s;
  int tid = threadIdx.x;
  int a = tid & 15, i = tid >> 4;

  float p = 0.f;
  for (int k = i; k < H; k += 16) p += h2[k] * soft_w[k * NA + a];
  red[tid] = p;
  __syncthreads();
  for (int s = 8; s > 0; s >>= 1) {
    if (i < s) red[tid] += red[tid + s * 16];
    __syncthreads();
  }

  if (tid == 0) {
    float logits[NA];
    for (int q = 0; q < NA; ++q) logits[q] = red[q];

    // step key: key(42)=(0,42); split(key,24) = threefry(key, iota(48)) -> (24,2)
    unsigned key0, key1, o0, o1;
    {
      unsigned m0 = 2u * (unsigned)step, m1 = m0 + 1u;
      unsigned p0 = m0 % 24u, p1 = m1 % 24u;
      threefry2x32(0u, 42u, p0, 24u + p0, o0, o1);
      key0 = (m0 < 24u) ? o0 : o1;
      threefry2x32(0u, 42u, p1, 24u + p1, o0, o1);
      key1 = (m1 < 24u) ? o0 : o1;
    }

    // 16 uniforms -> Gumbels: bits = threefry(key, iota(16)), pairs (j, 8+j)
    float gmb[NA];
    for (int pr = 0; pr < 8; ++pr) {
      threefry2x32(key0, key1, (unsigned)pr, (unsigned)(8 + pr), o0, o1);
      unsigned bs[2] = {o0, o1};
      #pragma unroll
      for (int hv = 0; hv < 2; ++hv) {
        float u = __uint_as_float((bs[hv] >> 9) | 0x3f800000u) - 1.0f;
        const float TINY = 1.1754944e-38f;
        u = u * (1.0f - TINY) + TINY;
        gmb[hv * 8 + pr] = -logf(-logf(u));
      }
    }

    int best = 0; float bv = logits[0] + gmb[0];
    for (int q = 1; q < NA; ++q) {
      float v = logits[q] + gmb[q];
      if (v > bv) { bv = v; best = q; }
    }

    float m = logits[0];
    for (int q = 1; q < NA; ++q) m = fmaxf(m, logits[q]);
    float se = 0.f, sl = 0.f;
    for (int q = 0; q < NA; ++q) {
      float e = expf(logits[q] - m);
      se += e; sl += e * logits[q];
    }
    float lse = m + logf(se);
    out[step]             = (float)best;           // action
    out[STEPS + step]     = logits[best] - lse;    // log-prob
    out[2 * STEPS + step] = lse - sl / se;         // entropy
    act_s = best;
  }
  __syncthreads();

  int act = act_s;
  for (int k = tid; k < H; k += 256)
    x_bf[k] = f32_to_bf16(w_emb[act * H + k]);
}

// ------------------------------- launcher -----------------------------------
extern "C" void kernel_launch(void* const* d_in, const int* in_sizes, int n_in,
                              void* d_out, int out_size, void* d_ws, size_t ws_size,
                              hipStream_t stream) {
  (void)in_sizes; (void)n_in; (void)out_size;
  const float* g_emb  = (const float*)d_in[0];
  const float* w_emb  = (const float*)d_in[1];
  const float* soft_w = (const float*)d_in[2];
  const float* w_ih   = (const float*)d_in[3];
  const float* w_hh   = (const float*)d_in[4];
  const float* b_ih   = (const float*)d_in[5];
  const float* b_hh   = (const float*)d_in[6];
  float* out = (float*)d_out;

  const size_t NW = (size_t)NL * G4H * H;                          // elems per weight tensor
  const size_t CACHE_BYTES = 2 * NW * sizeof(unsigned short);      // 134,217,728
  const size_t SMALL_BYTES = 512 * 1024;
  const bool cached = ws_size >= CACHE_BYTES + SMALL_BYTES;

  char* base = (char*)d_ws;
  unsigned short* wih_bf = (unsigned short*)base;
  unsigned short* whh_bf = wih_bf + NW;
  char* sb = cached ? (base + CACHE_BYTES) : base;
  unsigned short* x_bf       = (unsigned short*)(sb);              //  4 KB
  unsigned short* h_bf       = (unsigned short*)(sb + 4096);       //  8 KB
  float*          h_f        = (float*)(sb + 12288);               // 16 KB
  float*          c_f        = (float*)(sb + 28672);               // 16 KB
  float*          gates_part = (float*)(sb + 45056);               // 256 KB (KSPLIT*4H floats)

  if (cached) {
    long long n4 = (long long)(NW / 4);
    cvt_f32_bf16<<<2048, 256, 0, stream>>>(w_ih, wih_bf, n4);
    cvt_f32_bf16<<<2048, 256, 0, stream>>>(w_hh, whh_bf, n4);
  }
  init_state<<<H / 256, 256, 0, stream>>>(g_emb, x_bf, h_bf, h_f, c_f);

  dim3 ggrid(G4H / 256, KSPLIT);                    // (32, 8) -> 2048 waves
  for (int step = 0; step < STEPS; ++step) {
    for (int l = 0; l < NL; ++l) {
      const unsigned short* xb = (l == 0) ? x_bf : h_bf;   // layer1 input = layer0 h
      const unsigned short* hb = h_bf + l * H;
      size_t woff = (size_t)l * G4H * H;
      if (cached) {
        gemv_gates<true><<<ggrid, 256, 0, stream>>>(
            wih_bf + woff, whh_bf + woff, nullptr, nullptr, xb, hb, gates_part);
      } else {
        gemv_gates<false><<<ggrid, 256, 0, stream>>>(
            nullptr, nullptr, w_ih + woff, w_hh + woff, xb, hb, gates_part);
      }
      lstm_cell<<<H / 256, 256, 0, stream>>>(
          gates_part, b_ih + (size_t)l * G4H, b_hh + (size_t)l * G4H,
          c_f + l * H, h_f + l * H, h_bf + l * H);
    }
    sample_step<<<1, 256, 0, stream>>>(h_f + H, soft_w, w_emb, step, x_bf, out);
  }
}
